// CausalSelfAttention_8555574854367
// MI455X (gfx1250) — matrix operand order repaired and checked by reading, not compiled, not run on hardware
//
#include <hip/hip_runtime.h>
#include <hip/hip_bf16.h>

// ---------------- CDNA5 WMMA helpers ----------------
typedef __attribute__((ext_vector_type(8)))  float  v8f;
typedef __attribute__((ext_vector_type(16))) __bf16 v16bf;

union B16frag { uint4 u[2]; v16bf v; };

__device__ __forceinline__ v8f wmma_bf16(const B16frag& a, const B16frag& b, v8f c) {
  return __builtin_amdgcn_wmma_f32_16x16x32_bf16(false, a.v, false, b.v, (short)0, c,
                                                 false, false);
}

// ---------------- CDNA5 Tensor Data Mover (TDM) ----------------
typedef __attribute__((ext_vector_type(4))) unsigned int u32x4;
typedef __attribute__((ext_vector_type(8))) int          i32x8;
typedef __attribute__((ext_vector_type(4))) int          i32x4;

// 2D tile DMA: global (bf16 tensor, row stride `stride0` elements) -> LDS.
// tile_d0 = contiguous elements per row, tile_d1 = rows. LDS receives rows
// packed back-to-back (tile_d0*2 bytes each). Tracked by TENSORcnt.
__device__ __forceinline__ void tdm_load_2d(unsigned lds_addr, const void* gptr,
                                            unsigned tensor_d0, unsigned tensor_d1,
                                            unsigned tile_d0, unsigned tile_d1,
                                            unsigned long long stride0) {
  unsigned long long ga = (unsigned long long)(uintptr_t)gptr;
  u32x4 g0;
  g0[0] = 1u;                                                 // count=1 (valid), user mode
  g0[1] = lds_addr;                                           // LDS byte address
  g0[2] = (unsigned)(ga & 0xFFFFFFFFu);                       // global_addr[31:0]
  g0[3] = (unsigned)((ga >> 32) & 0x1FFFFFFu) | (2u << 30);   // global_addr[56:32], type=2
  i32x8 g1;
  g1[0] = (int)(1u << 16);                                    // data_size=1 (2 bytes)
  g1[1] = (int)((tensor_d0 & 0xFFFFu) << 16);                 // tensor_dim0[15:0]
  g1[2] = (int)(((tensor_d0 >> 16) & 0xFFFFu) |
                ((tensor_d1 & 0xFFFFu) << 16));               // dim0[31:16] | dim1[15:0]
  g1[3] = (int)(((tensor_d1 >> 16) & 0xFFFFu) |
                ((tile_d0 & 0xFFFFu) << 16));                 // dim1[31:16] | tile_dim0
  g1[4] = (int)(tile_d1 & 0xFFFFu);                           // tile_dim1 (tile_dim2=0)
  g1[5] = (int)(stride0 & 0xFFFFFFFFull);                     // tensor_dim0_stride[31:0]
  g1[6] = (int)((stride0 >> 32) & 0xFFFFull);                 // stride[47:32]
  g1[7] = 0;
  i32x4 z4 = {0, 0, 0, 0};
  i32x8 z8 = {0, 0, 0, 0, 0, 0, 0, 0};
  __builtin_amdgcn_tensor_load_to_lds(g0, g1, z4, z4, z8, 0);
}

__device__ __forceinline__ unsigned lds_byte_addr(const void* p) {
  // generic pointer to LDS: low 32 bits are the LDS byte address (ISA 10.2)
  return (unsigned)(uintptr_t)p;
}

// ---------------- problem constants ----------------
#define Bn   2
#define Tn   2048
#define Cn   2048
#define HQn  16
#define HKVn 4
#define Dn   128
#define Mn   (Bn * Tn)        // 4096
#define NQn  (HQn * Dn)       // 2048
#define NKVn (HKVn * Dn)      // 512

// ---------------- elementwise kernels ----------------
__global__ void cvt_f32_to_bf16(const float* __restrict__ src,
                                __hip_bfloat16* __restrict__ dst, int n) {
  int i = blockIdx.x * blockDim.x + threadIdx.x;
  if (i < n) dst[i] = __float2bfloat16(src[i]);
}

// Wt[n][k] = W[k][n]   (W is [K][N] row-major f32)
__global__ void transpose_w_bf16(const float* __restrict__ W,
                                 __hip_bfloat16* __restrict__ Wt, int K, int N) {
  int i = blockIdx.x * blockDim.x + threadIdx.x;
  int total = K * N;
  if (i >= total) return;
  int n = i / K, k = i - n * K;
  Wt[i] = __float2bfloat16(W[(size_t)k * N + n]);
}

// RoPE + rearrange: proj [B,T,H,d] (bf16) -> dst [B,H,T,d] (bf16)
__global__ void rope_rearrange(const __hip_bfloat16* __restrict__ proj,
                               __hip_bfloat16* __restrict__ dst, int H, int total) {
  int i = blockIdx.x * blockDim.x + threadIdx.x;
  if (i >= total) return;
  int j = i & (Dn - 1);
  int h = (i >> 7) % H;
  int t = (i / (Dn * H)) % Tn;
  int b = i / (Dn * H * Tn);
  float x = __bfloat162float(proj[i]);
  int jp = (j < Dn / 2) ? (j + Dn / 2) : (j - Dn / 2);
  float xr = __bfloat162float(proj[i - j + jp]);
  float rot = (j < Dn / 2) ? -xr : xr;
  float e = -(float)(2 * (j >> 1)) * (19.931568569324174f / 128.0f);  // log2(1e6)/d
  float ang = (float)t * exp2f(e);
  float cs, sn;
  __sincosf(ang, &sn, &cs);
  float val = x * cs + rot * sn;
  dst[(((size_t)(b * H + h) * Tn + t) << 7) + j] = __float2bfloat16(val);
}

// vproj [B,T,Hkv,d] -> vt [B,Hkv,d,T]
__global__ void v_transpose(const __hip_bfloat16* __restrict__ vproj,
                            __hip_bfloat16* __restrict__ vt, int total) {
  int i = blockIdx.x * blockDim.x + threadIdx.x;
  if (i >= total) return;
  int j  = i & (Dn - 1);
  int hk = (i >> 7) & (HKVn - 1);
  int t  = (i / (Dn * HKVn)) % Tn;
  int b  = i / (Dn * HKVn * Tn);
  vt[((size_t)(b * HKVn + hk) * Dn + j) * Tn + t] = vproj[i];
}

// ---------------- tiled WMMA GEMM (TDM-staged, double buffered) ----------------
// C[M,N] = A[M,K](bf16,row) @ Bt[N,K](bf16,row = B^T) + bias
// block: 128 threads = 4 waves; tile 64(M) x 128(N); K chunks of 32.
// Wave 0 drives the Tensor Data Mover; chunk k+1 streams in while k computes.
__global__ __launch_bounds__(128) void gemm_wmma_bf16(
    const __hip_bfloat16* __restrict__ A, const __hip_bfloat16* __restrict__ Bt,
    const float* __restrict__ bias, void* __restrict__ out,
    int M, int N, int K, int f32out) {
  const int lane = threadIdx.x & 31, wid = threadIdx.x >> 5;
  const int half = lane >> 4, ln = lane & 15;
  const int nb = blockIdx.x * 128, mb = blockIdx.y * 64;

  __shared__ __align__(16) __hip_bfloat16 As[2][64][32];    // 2 x 4 KB
  __shared__ __align__(16) __hip_bfloat16 Bs[2][128][32];   // 2 x 8 KB

  v8f acc[8] = {};

  const int nkc = K >> 5;
  const __hip_bfloat16* aBase = A + (size_t)mb * K;
  const __hip_bfloat16* bBase = Bt + (size_t)nb * K;

  if (wid == 0) {
    tdm_load_2d(lds_byte_addr(&As[0][0][0]), aBase, (unsigned)K, (unsigned)M, 32, 64,
                (unsigned long long)K);
    tdm_load_2d(lds_byte_addr(&Bs[0][0][0]), bBase, (unsigned)K, (unsigned)N, 32, 128,
                (unsigned long long)K);
  }

  for (int kc = 0; kc < nkc; ++kc) {
    const int cur = kc & 1;
    if (wid == 0) {
      if (kc + 1 < nkc) {
        const int nxt = cur ^ 1;
        tdm_load_2d(lds_byte_addr(&As[nxt][0][0]), aBase + (kc + 1) * 32,
                    (unsigned)K, (unsigned)M, 32, 64, (unsigned long long)K);
        tdm_load_2d(lds_byte_addr(&Bs[nxt][0][0]), bBase + (kc + 1) * 32,
                    (unsigned)K, (unsigned)N, 32, 128, (unsigned long long)K);
        __builtin_amdgcn_s_wait_tensorcnt(2);  // in-order: current buffer resident
      } else {
        __builtin_amdgcn_s_wait_tensorcnt(0);
      }
    }
    __syncthreads();

    B16frag a;
    a.u[0] = *reinterpret_cast<const uint4*>(&As[cur][wid * 16 + ln][half * 16]);
    a.u[1] = *reinterpret_cast<const uint4*>(&As[cur][wid * 16 + ln][half * 16 + 8]);
#pragma unroll
    for (int nt = 0; nt < 8; ++nt) {
      B16frag b;
      b.u[0] = *reinterpret_cast<const uint4*>(&Bs[cur][nt * 16 + ln][half * 16]);
      b.u[1] = *reinterpret_cast<const uint4*>(&Bs[cur][nt * 16 + ln][half * 16 + 8]);
      acc[nt] = wmma_bf16(a, b, acc[nt]);
    }
    __syncthreads();
  }

#pragma unroll
  for (int nt = 0; nt < 8; ++nt) {
#pragma unroll
    for (int r = 0; r < 8; ++r) {
      int col = nb + nt * 16 + ln;
      int row = mb + wid * 16 + r + 8 * half;
      float val = acc[nt][r] + (bias ? bias[col] : 0.0f);
      if (f32out)
        ((float*)out)[(size_t)row * N + col] = val;
      else
        ((__hip_bfloat16*)out)[(size_t)row * N + col] = __float2bfloat16(val);
    }
  }
}

// ---------------- flash attention (GQA, causal, RoPE'd inputs) ----------------
// qr [B,HQ,T,d], kr [B,HKV,T,d], vt [B,HKV,d,T]  -> yatt [B,T,HQ*d] (bf16)
// block: 128 threads = 4 waves; each wave owns 16 query rows; K/V 32-key tiles
// staged to LDS via the Tensor Data Mover (wave 0 drives TDM).
__global__ __launch_bounds__(128) void flash_attn_wmma(
    const __hip_bfloat16* __restrict__ qr, const __hip_bfloat16* __restrict__ kr,
    const __hip_bfloat16* __restrict__ vt, __hip_bfloat16* __restrict__ yatt) {
  const int lane = threadIdx.x & 31, wid = threadIdx.x >> 5;
  const int half = lane >> 4, ln = lane & 15;
  const int qt = blockIdx.x;       // 64-row q tile index
  const int h  = blockIdx.y;
  const int b  = blockIdx.z;
  const int hk = h >> 2;           // GQA group = 4
  const int qb = qt * 64 + wid * 16;

  __shared__ __align__(16) __hip_bfloat16 Ks[32][Dn];    // 8 KB
  __shared__ __align__(16) __hip_bfloat16 Vs[Dn][32];    // 8 KB (d-major)
  __shared__ __align__(16) __hip_bfloat16 Ps[4][16][32]; // 4 KB

  // Q fragments for this wave's 16 rows (A layout, 4 chunks of K=32)
  B16frag aq[4];
  const __hip_bfloat16* qrow = qr + (((size_t)(b * HQn + h) * Tn) + qb + ln) * Dn;
#pragma unroll
  for (int c = 0; c < 4; ++c) {
    aq[c].u[0] = *reinterpret_cast<const uint4*>(qrow + c * 32 + half * 16);
    aq[c].u[1] = *reinterpret_cast<const uint4*>(qrow + c * 32 + half * 16 + 8);
  }

  v8f o[8] = {};
  float mrow[8], lrow[8];
#pragma unroll
  for (int r = 0; r < 8; ++r) { mrow[r] = -3.0e38f; lrow[r] = 0.0f; }

  const __hip_bfloat16* ksrc = kr + ((size_t)(b * HKVn + hk) * Tn) * Dn;
  const __hip_bfloat16* vsrc = vt + ((size_t)(b * HKVn + hk) * Dn) * Tn;
  const float sc = 0.08838834764831845f;  // 1/sqrt(128)

  const int nkt = 2 * qt + 2;  // causal: keys < (qt+1)*64
  for (int it = 0; it < nkt; ++it) {
    const int kb = it * 32;
    __syncthreads();  // previous tile fully consumed before TDM overwrites
    if (wid == 0) {
      // K tile: 32 rows x 128 cols from [T][d]
      tdm_load_2d(lds_byte_addr(&Ks[0][0]), ksrc + (size_t)kb * Dn,
                  Dn, Tn, Dn, 32, (unsigned long long)Dn);
      // V tile: 128 rows x 32 cols from [d][T]
      tdm_load_2d(lds_byte_addr(&Vs[0][0]), vsrc + kb,
                  Tn, Dn, 32, Dn, (unsigned long long)Tn);
      __builtin_amdgcn_s_wait_tensorcnt(0);
    }
    __syncthreads();

    // S = Q @ K^T for 16 rows x 32 keys (two 16x16 tiles)
    v8f s0 = {}, s1 = {};
#pragma unroll
    for (int c = 0; c < 4; ++c) {
      B16frag b0, b1;
      b0.u[0] = *reinterpret_cast<const uint4*>(&Ks[ln][c * 32 + half * 16]);
      b0.u[1] = *reinterpret_cast<const uint4*>(&Ks[ln][c * 32 + half * 16 + 8]);
      b1.u[0] = *reinterpret_cast<const uint4*>(&Ks[16 + ln][c * 32 + half * 16]);
      b1.u[1] = *reinterpret_cast<const uint4*>(&Ks[16 + ln][c * 32 + half * 16 + 8]);
      s0 = wmma_bf16(aq[c], b0, s0);
      s1 = wmma_bf16(aq[c], b1, s1);
    }

    // online softmax per row (row M = r + 8*half lives in 16 lanes of this half)
#pragma unroll
    for (int r = 0; r < 8; ++r) {
      const int qrow_g = qb + r + 8 * half;
      const int k0 = kb + ln, k1 = kb + 16 + ln;
      float v0 = (k0 <= qrow_g) ? s0[r] * sc : -3.0e38f;
      float v1 = (k1 <= qrow_g) ? s1[r] * sc : -3.0e38f;
      float mx = fmaxf(v0, v1);
#pragma unroll
      for (int ofs = 8; ofs > 0; ofs >>= 1) mx = fmaxf(mx, __shfl_xor(mx, ofs, 32));
      const float mnew = fmaxf(mrow[r], mx);
      const float p0 = __expf(v0 - mnew);
      const float p1 = __expf(v1 - mnew);
      const float alpha = __expf(mrow[r] - mnew);
      float rs = p0 + p1;
#pragma unroll
      for (int ofs = 8; ofs > 0; ofs >>= 1) rs += __shfl_xor(rs, ofs, 32);
      lrow[r] = lrow[r] * alpha + rs;
      mrow[r] = mnew;
#pragma unroll
      for (int nt = 0; nt < 8; ++nt) o[nt][r] *= alpha;
      Ps[wid][r + 8 * half][ln]      = __float2bfloat16(p0);
      Ps[wid][r + 8 * half][16 + ln] = __float2bfloat16(p1);
    }

    // P (16x32) @ V (32x128): A-frag from per-wave Ps, B-frags from Vs (d-major)
    B16frag ap;
    ap.u[0] = *reinterpret_cast<const uint4*>(&Ps[wid][ln][half * 16]);
    ap.u[1] = *reinterpret_cast<const uint4*>(&Ps[wid][ln][half * 16 + 8]);
#pragma unroll
    for (int nt = 0; nt < 8; ++nt) {
      B16frag bv;
      bv.u[0] = *reinterpret_cast<const uint4*>(&Vs[nt * 16 + ln][half * 16]);
      bv.u[1] = *reinterpret_cast<const uint4*>(&Vs[nt * 16 + ln][half * 16 + 8]);
      o[nt] = wmma_bf16(ap, bv, o[nt]);
    }
  }

  // normalize and write y_att [B,T,HQ*d]
#pragma unroll
  for (int nt = 0; nt < 8; ++nt) {
#pragma unroll
    for (int r = 0; r < 8; ++r) {
      float val = o[nt][r] * (1.0f / lrow[r]);
      size_t row = (size_t)(b * Tn) + qb + r + 8 * half;
      yatt[row * (size_t)NQn + h * Dn + nt * 16 + ln] = __float2bfloat16(val);
    }
  }
}

// ---------------- host launcher ----------------
extern "C" void kernel_launch(void* const* d_in, const int* in_sizes, int n_in,
                              void* d_out, int out_size, void* d_ws, size_t ws_size,
                              hipStream_t stream) {
  (void)in_sizes; (void)n_in; (void)out_size; (void)ws_size;
  const float* x  = (const float*)d_in[0];
  const float* Wq = (const float*)d_in[1];
  const float* bq = (const float*)d_in[2];
  const float* Wk = (const float*)d_in[3];
  const float* bk = (const float*)d_in[4];
  const float* Wv = (const float*)d_in[5];
  const float* bv = (const float*)d_in[6];
  const float* Wo = (const float*)d_in[7];
  float* out = (float*)d_out;

  char* p = (char*)d_ws;
  size_t off = 0;
  auto take = [&](size_t bytes) { char* q = p + off; off += (bytes + 255) & ~(size_t)255; return q; };
  __hip_bfloat16* xb    = (__hip_bfloat16*)take((size_t)Mn * Cn * 2);     // 16 MB
  __hip_bfloat16* wqt   = (__hip_bfloat16*)take((size_t)Cn * NQn * 2);    // 8 MB
  __hip_bfloat16* wkt   = (__hip_bfloat16*)take((size_t)Cn * NKVn * 2);   // 2 MB
  __hip_bfloat16* wvt   = (__hip_bfloat16*)take((size_t)Cn * NKVn * 2);   // 2 MB
  __hip_bfloat16* wot   = (__hip_bfloat16*)take((size_t)NQn * Cn * 2);    // 8 MB
  __hip_bfloat16* qproj = (__hip_bfloat16*)take((size_t)Mn * NQn * 2);    // 16 MB
  __hip_bfloat16* kproj = (__hip_bfloat16*)take((size_t)Mn * NKVn * 2);   // 4 MB
  __hip_bfloat16* vproj = (__hip_bfloat16*)take((size_t)Mn * NKVn * 2);   // 4 MB
  __hip_bfloat16* kr    = (__hip_bfloat16*)take((size_t)Mn * NKVn * 2);   // 4 MB
  __hip_bfloat16* vtw   = (__hip_bfloat16*)take((size_t)Mn * NKVn * 2);   // 4 MB
  __hip_bfloat16* qr    = xb;     // alias: xb dead after projection GEMMs
  __hip_bfloat16* yatt  = qproj;  // alias: qproj dead after rope_q

  // 1) convert x, transpose+convert weights
  cvt_f32_to_bf16<<<(Mn * Cn) / 256, 256, 0, stream>>>(x, xb, Mn * Cn);
  transpose_w_bf16<<<(Cn * NQn) / 256, 256, 0, stream>>>(Wq, wqt, Cn, NQn);
  transpose_w_bf16<<<(Cn * NKVn) / 256, 256, 0, stream>>>(Wk, wkt, Cn, NKVn);
  transpose_w_bf16<<<(Cn * NKVn) / 256, 256, 0, stream>>>(Wv, wvt, Cn, NKVn);
  transpose_w_bf16<<<(NQn * Cn) / 256, 256, 0, stream>>>(Wo, wot, NQn, Cn);

  // 2) projection GEMMs (bf16 out, bias epilogue); N tiles of 128
  gemm_wmma_bf16<<<dim3(NQn / 128, Mn / 64), 128, 0, stream>>>(xb, wqt, bq, qproj, Mn, NQn, Cn, 0);
  gemm_wmma_bf16<<<dim3(NKVn / 128, Mn / 64), 128, 0, stream>>>(xb, wkt, bk, kproj, Mn, NKVn, Cn, 0);
  gemm_wmma_bf16<<<dim3(NKVn / 128, Mn / 64), 128, 0, stream>>>(xb, wvt, bv, vproj, Mn, NKVn, Cn, 0);

  // 3) RoPE + rearrange, V transpose
  rope_rearrange<<<(Mn * NQn) / 256, 256, 0, stream>>>(qproj, qr, HQn, Mn * NQn);
  rope_rearrange<<<(Mn * NKVn) / 256, 256, 0, stream>>>(kproj, kr, HKVn, Mn * NKVn);
  v_transpose<<<(Mn * NKVn) / 256, 256, 0, stream>>>(vproj, vtw, Mn * NKVn);

  // 4) flash attention (TDM-staged K/V tiles)
  flash_attn_wmma<<<dim3(Tn / 64, HQn, Bn), 128, 0, stream>>>(qr, kr, vtw, yatt);

  // 5) output projection -> f32
  gemm_wmma_bf16<<<dim3(Cn / 128, Mn / 64), 128, 0, stream>>>(yatt, wot, nullptr, out, Mn, Cn, NQn, 1);
}
